// DualChannelAttention_7928509628993
// MI455X (gfx1250) — compile-verified, matrix operand order checked by reading, not run on hardware
//
#include <hip/hip_runtime.h>
#include <hip/hip_bf16.h>

typedef unsigned short u16;
typedef __attribute__((ext_vector_type(16))) __bf16 v16bf;
typedef __attribute__((ext_vector_type(8)))  float  v8f;
typedef __attribute__((ext_vector_type(4)))  int    v4i;

union FragAB { v16bf v; uint4 q[2]; u16 u[16]; };
union FragC  { v8f   v; float f[8]; };

#if __has_builtin(__builtin_amdgcn_global_load_async_to_lds_b128) && \
    __has_builtin(__builtin_amdgcn_s_wait_asynccnt)
#define HAVE_ASYNC_LDS 1
typedef __attribute__((address_space(1))) v4i gv4i_t;   // global int4
typedef __attribute__((address_space(3))) v4i lv4i_t;   // LDS int4
#else
#define HAVE_ASYNC_LDS 0
#endif

__device__ __forceinline__ void stage_b128(const void* gsrc, void* ldst) {
#if HAVE_ASYNC_LDS
    __builtin_amdgcn_global_load_async_to_lds_b128(
        (gv4i_t*)(void*)gsrc, (lv4i_t*)ldst, 0, 0);
#else
    *(uint4*)ldst = *(const uint4*)gsrc;
#endif
}

__device__ __forceinline__ void stage_wait() {
#if HAVE_ASYNC_LDS
    __builtin_amdgcn_s_wait_asynccnt(0);
#endif
}

__device__ __forceinline__ u16 f2bf(float f) {
    unsigned int u = __builtin_bit_cast(unsigned int, f);
    u += 0x7FFFu + ((u >> 16) & 1u);           // round-to-nearest-even
    return (u16)(u >> 16);
}

// ---------------------------------------------------------------------------
// fp32 -> bf16 conversion
// ---------------------------------------------------------------------------
__global__ void cvt_bf16_kernel(const float* __restrict__ src, u16* __restrict__ dst, int n) {
    int i = blockIdx.x * blockDim.x + threadIdx.x;
    if (i < n) dst[i] = f2bf(src[i]);
}

// ---------------------------------------------------------------------------
// GEMM: out[M,N] = A[M,K](bf16) * W[K,N](fp32->bf16 staged transposed) + bias
// epi 0: store bf16 to outb ; epi 1: store fp32 to outf ; epi 2: outf += wc*val
// WG: 256 threads = 8 waves (4x2), tile 128(M) x 64(N), 32x32 per wave,
// K step 32 -> 4 WMMA per wave per step, all fragments loaded as 2x b128.
// A tile staged via async global->LDS (ASYNCcnt); W converted through VGPRs.
// ---------------------------------------------------------------------------
#define GMT 128
#define GNT 64
#define GKT 32

__global__ __launch_bounds__(256) void gemm_wmma_kernel(
    const u16* __restrict__ A, const float* __restrict__ W,
    const float* __restrict__ bias, u16* __restrict__ outb,
    float* __restrict__ outf, const float* __restrict__ wcross,
    int M, int K, int N, int epi)
{
    __shared__ __align__(16) u16 As [GMT][GKT + 8];   // row-major, frag-contiguous
    __shared__ __align__(16) u16 BsT[GNT][GKT + 8];   // W^T: [n][k], frag-contiguous

    const int tid  = threadIdx.x;
    const int wave = tid >> 5;
    const int lane = tid & 31;
    const int half = lane >> 4;
    const int lr   = lane & 15;

    const int n0 = blockIdx.x * GNT;
    const int m0 = blockIdx.y * GMT;
    const int mw = (wave >> 1) << 5;   // 0,32,64,96
    const int nw = (wave & 1) << 5;    // 0,32

    FragC c[2][2];
#pragma unroll
    for (int r = 0; r < 2; ++r)
#pragma unroll
        for (int n = 0; n < 2; ++n)
#pragma unroll
            for (int j = 0; j < 8; ++j) c[r][n].f[j] = 0.0f;

    const int nk = K / GKT;
    for (int kb = 0; kb < nk; ++kb) {
        const int k0 = kb * GKT;
        // stage A: 128 rows x 4 chunks of 8 bf16 = 512 x b128, 2 per thread
#pragma unroll
        for (int it = 0; it < 2; ++it) {
            int cch = tid + it * 256;
            int row = cch >> 2, off = (cch & 3) * 8;
            stage_b128(&A[(size_t)(m0 + row) * K + (k0 + off)], &As[row][off]);
        }
        // stage W transposed: 32 k-rows x 16 float4 chunks, 2 per thread
#pragma unroll
        for (int it = 0; it < 2; ++it) {
            int cch = tid + it * 256;
            int kr = cch >> 4, nc = (cch & 15) * 4;
            float4 wv = *(const float4*)&W[(size_t)(k0 + kr) * N + (n0 + nc)];
            BsT[nc + 0][kr] = f2bf(wv.x);
            BsT[nc + 1][kr] = f2bf(wv.y);
            BsT[nc + 2][kr] = f2bf(wv.z);
            BsT[nc + 3][kr] = f2bf(wv.w);
            // speculative prefetch of next K-step (safe past end: silently dropped)
            __builtin_prefetch(&W[(size_t)(k0 + GKT + kr) * N + (n0 + nc)], 0, 1);
        }
        stage_wait();
        __syncthreads();

        FragAB a[2], b[2];
#pragma unroll
        for (int r = 0; r < 2; ++r) {
            a[r].q[0] = *(const uint4*)&As[mw + r * 16 + lr][half * 8];
            a[r].q[1] = *(const uint4*)&As[mw + r * 16 + lr][16 + half * 8];
        }
#pragma unroll
        for (int n = 0; n < 2; ++n) {
            b[n].q[0] = *(const uint4*)&BsT[nw + n * 16 + lr][half * 8];
            b[n].q[1] = *(const uint4*)&BsT[nw + n * 16 + lr][16 + half * 8];
        }
#pragma unroll
        for (int r = 0; r < 2; ++r)
#pragma unroll
            for (int n = 0; n < 2; ++n)
                c[r][n].v = __builtin_amdgcn_wmma_f32_16x16x32_bf16(
                    false, a[r].v, false, b[n].v, (short)0, c[r][n].v, false, false);
        __syncthreads();
    }

#pragma unroll
    for (int n = 0; n < 2; ++n) {
        const int col = n0 + nw + n * 16 + lr;
        const float bv = bias[col];
#pragma unroll
        for (int r = 0; r < 2; ++r)
#pragma unroll
            for (int v = 0; v < 8; ++v) {
                const int rr = m0 + mw + r * 16 + v + half * 8;
                const float val = c[r][n].f[v] + bv;
                if (epi == 0) {
                    outb[(size_t)rr * N + col] = f2bf(val);
                } else if (epi == 1) {
                    outf[(size_t)rr * N + col] = val;
                } else {
                    outf[(size_t)rr * N + col] += wcross[0] * val;
                }
            }
    }
}

// ---------------------------------------------------------------------------
// Flash attention, bf16 WMMA, online softmax.
// Block: 128 threads = 4 waves, each wave owns 16 query rows (64 q / block).
// Key loop step 32. DH = head dim (64 primary, 4 secondary; padded in LDS).
// Q/K staged async global->LDS; V staged transposed for contiguous B-frags.
// ---------------------------------------------------------------------------
#define FA_S 2048

template <int DH>
__global__ __launch_bounds__(128) void flash_attn_kernel(
    const u16* __restrict__ Q, const u16* __restrict__ Kb,
    const u16* __restrict__ Vb, u16* __restrict__ O, float scale)
{
    constexpr int DHP  = (DH >= 32) ? DH : 32;   // padded feature dim for QK^T
    constexpr int VN   = (DH >= 16) ? DH : 16;   // padded value/output cols
    constexpr int DTOT = 16 * DH;                // H * DH
    constexpr int KS   = DHP / 32;               // WMMA K-steps for QK^T
    constexpr int NT   = VN / 16;                // output col tiles

    __shared__ __align__(16) u16 Qs [64][DHP];
    __shared__ __align__(16) u16 Ks [32][DHP];
    __shared__ __align__(16) u16 VsT[VN][32];    // transposed: [feature][key]
    __shared__ __align__(16) u16 Ps [4][16 * 32];

    const int tid  = threadIdx.x;
    const int w    = tid >> 5;
    const int lane = tid & 31;
    const int half = lane >> 4;
    const int lr   = lane & 15;

    const int q0 = blockIdx.x * 64;
    const int h  = blockIdx.y;
    const int b  = blockIdx.z;
    const size_t headoff = (size_t)h * DH;

    // stage Q
    if constexpr (DH == DHP) {
#pragma unroll
        for (int it = 0; it < 4; ++it) {
            int cch = tid + it * 128;
            int r = cch >> 3, off = (cch & 7) * 8;
            stage_b128(&Q[(size_t)(b * FA_S + q0 + r) * DTOT + headoff + off],
                       &Qs[r][off]);
        }
        stage_wait();
    } else {
        for (int i = tid; i < 64 * DHP; i += 128) {
            int r = i / DHP, cc = i % DHP;
            Qs[r][cc] = (cc < DH)
                ? Q[(size_t)(b * FA_S + q0 + r) * DTOT + headoff + cc] : (u16)0;
        }
    }
    __syncthreads();

    FragAB aq[KS];
#pragma unroll
    for (int ks = 0; ks < KS; ++ks) {
        aq[ks].q[0] = *(const uint4*)&Qs[w * 16 + lr][ks * 32 + half * 8];
        aq[ks].q[1] = *(const uint4*)&Qs[w * 16 + lr][ks * 32 + 16 + half * 8];
    }

    float mrow[8], lrow[8];
    FragC oacc[NT];
#pragma unroll
    for (int v = 0; v < 8; ++v) { mrow[v] = -3.0e38f; lrow[v] = 0.0f; }
#pragma unroll
    for (int n = 0; n < NT; ++n)
#pragma unroll
        for (int v = 0; v < 8; ++v) oacc[n].f[v] = 0.0f;

    for (int kb = 0; kb < FA_S / 32; ++kb) {
        __syncthreads();
        // stage K block (row-major, async) and V block (transposed)
        if constexpr (DH == DHP) {
#pragma unroll
            for (int it = 0; it < 2; ++it) {
                int cch = tid + it * 128;
                int r = cch >> 3, off = (cch & 7) * 8;
                stage_b128(&Kb[(size_t)(b * FA_S + kb * 32 + r) * DTOT + headoff + off],
                           &Ks[r][off]);
            }
#pragma unroll
            for (int it = 0; it < 2; ++it) {
                int cch = tid + it * 128;
                int r = cch >> 3, off = (cch & 7) * 8;
                FragAB tmp;
                tmp.q[0] = *(const uint4*)&Vb[(size_t)(b * FA_S + kb * 32 + r) * DTOT + headoff + off];
#pragma unroll
                for (int j = 0; j < 8; ++j) VsT[off + j][r] = tmp.u[j];
            }
            stage_wait();
        } else {
            for (int i = tid; i < 32 * DHP; i += 128) {
                int r = i / DHP, cc = i % DHP;
                Ks[r][cc] = (cc < DH)
                    ? Kb[(size_t)(b * FA_S + kb * 32 + r) * DTOT + headoff + cc] : (u16)0;
            }
            for (int i = tid; i < 32 * VN; i += 128) {
                int r = i / VN, cc = i % VN;
                VsT[cc][r] = (cc < DH)
                    ? Vb[(size_t)(b * FA_S + kb * 32 + r) * DTOT + headoff + cc] : (u16)0;
            }
        }
        __syncthreads();

        // S = scale * Q K^T   (two 16x16 key tiles = 32 keys)
        FragC sc[2];
#pragma unroll
        for (int t = 0; t < 2; ++t) {
#pragma unroll
            for (int v = 0; v < 8; ++v) sc[t].f[v] = 0.0f;
#pragma unroll
            for (int ks = 0; ks < KS; ++ks) {
                FragAB bk;
                bk.q[0] = *(const uint4*)&Ks[t * 16 + lr][ks * 32 + half * 8];
                bk.q[1] = *(const uint4*)&Ks[t * 16 + lr][ks * 32 + 16 + half * 8];
                sc[t].v = __builtin_amdgcn_wmma_f32_16x16x32_bf16(
                    false, aq[ks].v, false, bk.v, (short)0, sc[t].v, false, false);
            }
#pragma unroll
            for (int v = 0; v < 8; ++v) sc[t].f[v] *= scale;
        }

        // online softmax: row reductions across the 16-lane half
        float rm[8], rs[8], al[8];
#pragma unroll
        for (int v = 0; v < 8; ++v) rm[v] = fmaxf(sc[0].f[v], sc[1].f[v]);
#pragma unroll
        for (int off = 1; off <= 8; off <<= 1)
#pragma unroll
            for (int v = 0; v < 8; ++v)
                rm[v] = fmaxf(rm[v], __shfl_xor(rm[v], off, 32));
#pragma unroll
        for (int v = 0; v < 8; ++v) {
            float mn = fmaxf(mrow[v], rm[v]);
            al[v] = __expf(mrow[v] - mn);
            mrow[v] = mn;
            float p0 = __expf(sc[0].f[v] - mn);
            float p1 = __expf(sc[1].f[v] - mn);
            sc[0].f[v] = p0; sc[1].f[v] = p1;
            rs[v] = p0 + p1;
        }
#pragma unroll
        for (int off = 1; off <= 8; off <<= 1)
#pragma unroll
            for (int v = 0; v < 8; ++v)
                rs[v] += __shfl_xor(rs[v], off, 32);
#pragma unroll
        for (int v = 0; v < 8; ++v) lrow[v] = lrow[v] * al[v] + rs[v];
#pragma unroll
        for (int n = 0; n < NT; ++n)
#pragma unroll
            for (int v = 0; v < 8; ++v) oacc[n].f[v] *= al[v];

        // re-layout P (C-layout -> A-layout) through this wave's LDS strip
#pragma unroll
        for (int t = 0; t < 2; ++t)
#pragma unroll
            for (int v = 0; v < 8; ++v)
                Ps[w][(v + 8 * half) * 32 + t * 16 + lr] = f2bf(sc[t].f[v]);
        __syncthreads();

        FragAB ap;
        ap.q[0] = *(const uint4*)&Ps[w][lr * 32 + half * 8];
        ap.q[1] = *(const uint4*)&Ps[w][lr * 32 + 16 + half * 8];
#pragma unroll
        for (int n = 0; n < NT; ++n) {
            FragAB bv;
            bv.q[0] = *(const uint4*)&VsT[n * 16 + lr][half * 8];
            bv.q[1] = *(const uint4*)&VsT[n * 16 + lr][16 + half * 8];
            oacc[n].v = __builtin_amdgcn_wmma_f32_16x16x32_bf16(
                false, ap.v, false, bv.v, (short)0, oacc[n].v, false, false);
        }
    }

    // normalize + store
#pragma unroll
    for (int n = 0; n < NT; ++n) {
        const int col = n * 16 + lr;
        if (col < DH) {
#pragma unroll
            for (int v = 0; v < 8; ++v) {
                const int rr = q0 + w * 16 + v + 8 * half;
                O[(size_t)(b * FA_S + rr) * DTOT + headoff + col] =
                    f2bf(oacc[n].f[v] / lrow[v]);
            }
        }
    }
}

// ---------------------------------------------------------------------------
// launch
// ---------------------------------------------------------------------------
extern "C" void kernel_launch(void* const* d_in, const int* in_sizes, int n_in,
                              void* d_out, int out_size, void* d_ws, size_t ws_size,
                              hipStream_t stream) {
    (void)in_sizes; (void)n_in; (void)out_size; (void)ws_size;
    const int Bn = 2, S = 2048, D = 1024, H = 16, HD = 64, R = 4;
    const int M = Bn * S;            // 4096
    const int Ds = H * R;            // 64

    const float* x    = (const float*)d_in[0];
    const float* wq_p = (const float*)d_in[1];  const float* bq_p = (const float*)d_in[2];
    const float* wk_p = (const float*)d_in[3];  const float* bk_p = (const float*)d_in[4];
    const float* wv_p = (const float*)d_in[5];  const float* bv_p = (const float*)d_in[6];
    const float* wo_p = (const float*)d_in[7];  const float* bo_p = (const float*)d_in[8];
    const float* wq_s = (const float*)d_in[9];  const float* bq_s = (const float*)d_in[10];
    const float* wk_s = (const float*)d_in[11]; const float* bk_s = (const float*)d_in[12];
    const float* wv_s = (const float*)d_in[13]; const float* bv_s = (const float*)d_in[14];
    const float* wo_s = (const float*)d_in[15]; const float* bo_s = (const float*)d_in[16];
    const float* w_cross = (const float*)d_in[17];
    float* out = (float*)d_out;

    char* ws = (char*)d_ws;
    size_t off = 0;
    u16* xb  = (u16*)(ws + off); off += (size_t)M * D * 2;
    u16* Qp  = (u16*)(ws + off); off += (size_t)M * D * 2;
    u16* Kp  = (u16*)(ws + off); off += (size_t)M * D * 2;
    u16* Vp  = (u16*)(ws + off); off += (size_t)M * D * 2;
    u16* Ap  = (u16*)(ws + off); off += (size_t)M * D * 2;
    u16* Qs2 = (u16*)(ws + off); off += (size_t)M * Ds * 2;
    u16* Ks2 = (u16*)(ws + off); off += (size_t)M * Ds * 2;
    u16* Vs2 = (u16*)(ws + off); off += (size_t)M * Ds * 2;
    u16* As2 = (u16*)(ws + off); off += (size_t)M * Ds * 2;

    // 1) x -> bf16
    {
        int n = M * D;
        cvt_bf16_kernel<<<(n + 255) / 256, 256, 0, stream>>>(x, xb, n);
    }
    // 2) QKV projections (primary D->D, secondary D->64)
    dim3 gP(D / GNT, M / GMT), gS(Ds / GNT, M / GMT);
    gemm_wmma_kernel<<<gP, 256, 0, stream>>>(xb, wq_p, bq_p, Qp, nullptr, nullptr, M, D, D, 0);
    gemm_wmma_kernel<<<gP, 256, 0, stream>>>(xb, wk_p, bk_p, Kp, nullptr, nullptr, M, D, D, 0);
    gemm_wmma_kernel<<<gP, 256, 0, stream>>>(xb, wv_p, bv_p, Vp, nullptr, nullptr, M, D, D, 0);
    gemm_wmma_kernel<<<gS, 256, 0, stream>>>(xb, wq_s, bq_s, Qs2, nullptr, nullptr, M, D, Ds, 0);
    gemm_wmma_kernel<<<gS, 256, 0, stream>>>(xb, wk_s, bk_s, Ks2, nullptr, nullptr, M, D, Ds, 0);
    gemm_wmma_kernel<<<gS, 256, 0, stream>>>(xb, wv_s, bv_s, Vs2, nullptr, nullptr, M, D, Ds, 0);
    // 3) flash attention
    dim3 gf(S / 64, H, Bn);
    flash_attn_kernel<HD><<<gf, 128, 0, stream>>>(Qp, Kp, Vp, Ap, 0.125f);  // 1/sqrt(64)
    flash_attn_kernel<R><<<gf, 128, 0, stream>>>(Qs2, Ks2, Vs2, As2, 0.5f); // 1/sqrt(4)
    // 4) output projections: out = Ap@wo_p + bo_p; out += w_cross*(As2@wo_s + bo_s)
    gemm_wmma_kernel<<<gP, 256, 0, stream>>>(Ap, wo_p, bo_p, nullptr, out, nullptr, M, D, D, 1);
    gemm_wmma_kernel<<<gP, 256, 0, stream>>>(As2, wo_s, bo_s, nullptr, out, w_cross, M, Ds, D, 2);
}
